// hyp_GCN_7997229105402
// MI455X (gfx1250) — compile-verified, matrix operand order checked by reading, not run on hardware
//
#include <hip/hip_runtime.h>
#include <math.h>
#include <stdint.h>

// ---------------------------------------------------------------------------
// Hyperbolic GCN + HGP-SL pooling forward, MI455X (gfx1250, wave32, WMMA).
// Block-diagonal adjacency (32 graphs) -> batched 16x16-tile WMMA GEMMs.
// GEMM data path: async global->LDS B128 copies (ASYNCcnt) for the fp32 A
// slab, f16 transpose-staged B tile, fragments assembled from LDS with packed
// loads, v_wmma_f32_16x16x32_f16 compute. Working set (~30MB) lives in L2.
// ---------------------------------------------------------------------------

#define NB    32      // graphs
#define NPG0  256     // nodes per graph, layer 1
#define NTOT  (NB*NPG0)
#define DIN   128     // input feature dim
#define DH    64      // hidden dim
#define MINN  1e-15f
#define MAXN  (1.0f - 4e-3f)

typedef __attribute__((ext_vector_type(16))) _Float16 v16h;
typedef __attribute__((ext_vector_type(8)))  float    v8f;

__device__ __forceinline__ float wave_sum(float v) {
#pragma unroll
  for (int off = 16; off > 0; off >>= 1) v += __shfl_xor(v, off, 32);
  return v;
}

__device__ __forceinline__ float artanhf_(float x) {
  x = fminf(fmaxf(x, -1.0f + 1e-7f), 1.0f - 1e-7f);
  return 0.5f * (log1pf(x) - log1pf(-x));
}

// async global->LDS 128-bit copy; LDS offset = low 32 bits of the generic
// pointer (flat LDS aperture keeps the byte offset in addr[31:0]).
__device__ __forceinline__ void async_copy_b128(void* lds_dst, const void* gsrc) {
  unsigned dst = (unsigned)(uintptr_t)lds_dst;
  unsigned long long src = (unsigned long long)(uintptr_t)gsrc;
  asm volatile("global_load_async_to_lds_b128 %0, %1, off"
               :: "v"(dst), "v"(src) : "memory");
}
__device__ __forceinline__ void wait_async0() {
  asm volatile("s_wait_asynccnt 0" ::: "memory");
}

// ---------------------------------------------------------------------------
// WMMA GEMM: C[M,N] = A[M,K] * B  (BT: B given as [N,K] row-major, i.e. W;
// else B is [K,N] row-major). Block = 128 threads = 4 waves; each wave owns
// one 16x16 C tile of a 64x16 block tile. blockIdx.z batches block-diagonal
// operands. K multiple of 32; M multiple of 64; N multiple of 16.
// A fragment (16x32 f16, wave32): lane l holds row (l&15), k-half by l>>4.
// B fragment mirrors with lane = column. C/D: VGPR r, lane l ->
// (M = r + 8*(l>>4), N = l&15).
// ---------------------------------------------------------------------------
template <bool BT>
__global__ __launch_bounds__(128)
void wmma_gemm(const float* __restrict__ A, const float* __restrict__ B,
               float* __restrict__ C, int M, int N, int K,
               long long sA, long long sB, long long sC) {
  __shared__ __align__(16) float    Ash[64 * 32];  // fp32 A slab (async DMA)
  __shared__ __align__(16) _Float16 Bsh[16 * 32];  // f16 B tile, [n][k]
  const int tid  = threadIdx.x;
  const int wave = tid >> 5;
  const int lane = tid & 31;
  const int hi   = lane >> 4;
  const int lo   = lane & 15;
  const int tm   = blockIdx.x * 64;
  const int tn   = blockIdx.y * 16;
  A += (long long)blockIdx.z * sA;
  B += (long long)blockIdx.z * sB;
  C += (long long)blockIdx.z * sC;

  v8f acc = {};
  for (int k0 = 0; k0 < K; k0 += 32) {
    __syncthreads();  // protect LDS reuse across K-slabs
    // ---- A slab 64x32 fp32 -> LDS via async b128 copies (ASYNCcnt) ----
#pragma unroll
    for (int i = 0; i < 4; ++i) {
      int idx = i * 128 + tid;          // 512 float4's
      int row = idx >> 3, c4 = idx & 7; // 8 float4 per 32-float row
      async_copy_b128(&Ash[row * 32 + c4 * 4],
                      A + (long long)(tm + row) * K + k0 + c4 * 4);
    }
    if (k0 + 32 < K) {  // speculative prefetch of next slab into L2
      __builtin_prefetch(&A[(long long)(tm + (tid >> 1)) * K + k0 + 32], 0, 0);
    }
    // ---- B tile -> f16 LDS, k-contiguous per output column ----
#pragma unroll
    for (int i = 0; i < 4; ++i) {
      int idx = i * 128 + tid;          // 512 elements
      int n, k;
      if (BT) { n = idx >> 5; k = idx & 31; }
      else    { k = idx >> 4; n = idx & 15; }
      float v = BT ? B[(long long)(tn + n) * K + k0 + k]
                   : B[(long long)(k0 + k) * N + tn + n];
      Bsh[n * 32 + k] = (_Float16)v;
    }
    wait_async0();
    __syncthreads();
    // ---- fragments from LDS (packed pair reads + cvt_pk) ----
    v16h a, b;
#pragma unroll
    for (int p = 0; p < 8; ++p) {
      const int kk = ((p & 4) << 2) + ((p & 3) << 1) + (hi << 3);
      const float* ap = &Ash[(wave * 16 + lo) * 32 + kk];
      a[2 * p]     = (_Float16)ap[0];
      a[2 * p + 1] = (_Float16)ap[1];
      const _Float16* bp = &Bsh[lo * 32 + kk];
      b[2 * p]     = bp[0];
      b[2 * p + 1] = bp[1];
    }
    acc = __builtin_amdgcn_wmma_f32_16x16x32_f16(false, a, false, b,
                                                 (short)0, acc, false, false);
  }
#pragma unroll
  for (int r = 0; r < 8; ++r)
    C[(long long)(tm + wave * 16 + r + hi * 8) * N + (tn + lo)] = acc[r];
}

// --------------------------- adjacency build -------------------------------
__global__ void k_build_adj(const float* __restrict__ x, const int* __restrict__ ei,
                            float* __restrict__ adj, int E, int npg, int Din) {
  int e = blockIdx.x * blockDim.x + threadIdx.x;
  if (e >= E) return;
  int r = ei[e], c = ei[E + e];
  const float4* xr = (const float4*)(x + (long long)r * Din);
  const float4* xc = (const float4*)(x + (long long)c * Din);
  float s = 0.f;
  for (int j = 0; j < Din / 4; ++j) {
    float4 a = xr[j], b = xc[j];
    s += a.x + a.y + a.z + a.w + b.x + b.y + b.z + b.w;
  }
  int g = r / npg, rl = r - g * npg, cl = c - g * npg;
  adj[((long long)g * npg + rl) * npg + cl] = 0.5f * s;
}

// -------------------- row-wise hyperbolic elementwise ops ------------------
__global__ void k_expmap0_proj(const float* __restrict__ in, float* __restrict__ out,
                               int nrows, int D) {
  int row = blockIdx.x * (blockDim.x >> 5) + (threadIdx.x >> 5);
  int lane = threadIdx.x & 31;
  if (row >= nrows) return;
  const float* x = in + (long long)row * D;
  float ss = 0.f;
  for (int j = lane; j < D; j += 32) { float v = x[j]; ss += v * v; }
  ss = wave_sum(ss);
  float n = fmaxf(sqrtf(ss), MINN);
  float tn = tanhf(n);
  float s = (tn > MAXN) ? (MAXN / n) : (tn / n);
  float* o = out + (long long)row * D;
  for (int j = lane; j < D; j += 32) o[j] = x[j] * s;
}

__global__ void k_logmap0(const float* __restrict__ in, float* __restrict__ out,
                          int nrows) {
  int row = blockIdx.x * (blockDim.x >> 5) + (threadIdx.x >> 5);
  int lane = threadIdx.x & 31;
  if (row >= nrows) return;
  const float* x = in + (long long)row * DH;
  float x0 = x[lane], x1 = x[lane + 32];
  float n = fmaxf(sqrtf(wave_sum(x0 * x0 + x1 * x1)), MINN);
  float s = artanhf_(n) / n;
  float* o = out + (long long)row * DH;
  o[lane] = x0 * s; o[lane + 32] = x1 * s;
}

__global__ void k_hyp_bias(const float* __restrict__ b, float* __restrict__ hb) {
  int lane = threadIdx.x;
  float v0 = b[lane], v1 = b[lane + 32];
  float n = fmaxf(sqrtf(wave_sum(v0 * v0 + v1 * v1)), MINN);
  float tn = tanhf(n);
  float s = (tn > MAXN) ? (MAXN / n) : (tn / n);
  hb[lane] = v0 * s; hb[lane + 32] = v1 * s;
}

__global__ void k_mobius(const float* __restrict__ xh, const float* __restrict__ mx,
                         const float* __restrict__ hb, float* __restrict__ out,
                         int nrows, int Din) {
  int row = blockIdx.x * (blockDim.x >> 5) + (threadIdx.x >> 5);
  int lane = threadIdx.x & 31;
  if (row >= nrows) return;
  const float* xr = xh + (long long)row * Din;
  float ss = 0.f;
  for (int j = lane; j < Din; j += 32) { float v = xr[j]; ss += v * v; }
  float xn = fmaxf(sqrtf(wave_sum(ss)), MINN);
  const float* mr = mx + (long long)row * DH;
  float m0 = mr[lane], m1 = mr[lane + 32];
  float mxn = fmaxf(sqrtf(wave_sum(m0 * m0 + m1 * m1)), MINN);
  float r = tanhf(mxn / xn * artanhf_(xn)) / mxn;
  float h0 = m0 * r, h1 = m1 * r;
  float hn = fmaxf(sqrtf(wave_sum(h0 * h0 + h1 * h1)), MINN);
  if (hn > MAXN) { float s = MAXN / hn; h0 *= s; h1 *= s; }
  float y0 = hb[lane], y1 = hb[lane + 32];
  float x2 = wave_sum(h0 * h0 + h1 * h1);
  float y2 = wave_sum(y0 * y0 + y1 * y1);
  float xy = wave_sum(h0 * y0 + h1 * y1);
  float ca = 1.f + 2.f * xy + y2;
  float cb = 1.f - x2;
  float den = fmaxf(1.f + 2.f * xy + x2 * y2, MINN);
  float z0 = (ca * h0 + cb * y0) / den;
  float z1 = (ca * h1 + cb * y1) / den;
  float zn = fmaxf(sqrtf(wave_sum(z0 * z0 + z1 * z1)), MINN);
  if (zn > MAXN) { float s = MAXN / zn; z0 *= s; z1 *= s; }
  float* o = out + (long long)row * DH;
  o[lane] = z0; o[lane + 32] = z1;
}

__global__ void k_logrelu(const float* __restrict__ in, float* __restrict__ out,
                          int nrows) {
  int row = blockIdx.x * (blockDim.x >> 5) + (threadIdx.x >> 5);
  int lane = threadIdx.x & 31;
  if (row >= nrows) return;
  const float* h = in + (long long)row * DH;
  float h0 = h[lane], h1 = h[lane + 32];
  float hn = fmaxf(sqrtf(wave_sum(h0 * h0 + h1 * h1)), MINN);
  float ls = artanhf_(hn) / hn;
  float u0 = fmaxf(h0 * ls, 0.f), u1 = fmaxf(h1 * ls, 0.f);
  float un = fmaxf(sqrtf(wave_sum(u0 * u0 + u1 * u1)), MINN);
  float tn = tanhf(un);
  float s = (tn > MAXN) ? (MAXN / un) : (tn / un);
  float y0 = u0 * s, y1 = u1 * s;
  float yn = fmaxf(sqrtf(wave_sum(y0 * y0 + y1 * y1)), MINN);
  float s2 = artanhf_(yn) / yn;
  float* o = out + (long long)row * DH;
  o[lane] = y0 * s2; o[lane + 32] = y1 * s2;
}

// ------------------------------ pooling ------------------------------------
__global__ void k_degree(const float* __restrict__ adj, float* __restrict__ dis,
                         int nrows, int npg) {
  int row = blockIdx.x * (blockDim.x >> 5) + (threadIdx.x >> 5);
  int lane = threadIdx.x & 31;
  if (row >= nrows) return;
  int g = row / npg, r = row - g * npg;
  const float* a = adj + ((long long)g * npg + r) * npg;
  float s = 0.f;
  for (int j = lane; j < npg; j += 32) s += a[j];
  s = wave_sum(s);
  if (lane == 0) dis[row] = (s > 0.f) ? rsqrtf(s) : 0.f;
}

__global__ void k_scale_rows(const float* __restrict__ in, const float* __restrict__ dis,
                             float* __restrict__ out, int n) {
  int i = blockIdx.x * blockDim.x + threadIdx.x;
  if (i >= n) return;
  out[i] = dis[i >> 6] * in[i];
}

__global__ void k_score(const float* __restrict__ t, const float* __restrict__ prop,
                        const float* __restrict__ dis, float* __restrict__ score,
                        int nrows) {
  int row = blockIdx.x * (blockDim.x >> 5) + (threadIdx.x >> 5);
  int lane = threadIdx.x & 31;
  if (row >= nrows) return;
  const float* tr = t + (long long)row * DH;
  const float* pr = prop + (long long)row * DH;
  float d = dis[row];
  float s = fabsf(tr[lane] - d * pr[lane]) + fabsf(tr[lane + 32] - d * pr[lane + 32]);
  s = wave_sum(s);
  if (lane == 0) score[row] = s;
}

__global__ void k_topk(const float* __restrict__ score, int* __restrict__ gidx,
                       float* __restrict__ vals, int npg, int k) {
  __shared__ float sc[256];
  __shared__ int   id[256];
  int g = blockIdx.x, t = threadIdx.x;
  sc[t] = score[g * npg + t];
  id[t] = t;
  __syncthreads();
  for (int size = 2; size <= npg; size <<= 1) {
    for (int stride = size >> 1; stride > 0; stride >>= 1) {
      int j = t ^ stride;
      if (j > t) {
        bool desc = ((t & size) == 0);
        float si = sc[t], sj = sc[j];
        int ii = id[t], ij = id[j];
        bool sw = desc ? (si < sj || (si == sj && ii > ij))
                       : (si > sj || (si == sj && ii < ij));
        if (sw) { sc[t] = sj; sc[j] = si; id[t] = ij; id[j] = ii; }
      }
      __syncthreads();
    }
  }
  if (t < k) {
    gidx[g * k + t] = g * npg + id[t];
    vals[g * k + t] = sc[t];
  }
}

__global__ void k_gather_scale(const float* __restrict__ tin, const int* __restrict__ gidx,
                               const float* __restrict__ vals, float* __restrict__ out,
                               int n) {
  int idx = blockIdx.x * blockDim.x + threadIdx.x;
  if (idx >= n) return;
  int i = idx >> 6, f = idx & 63;
  out[idx] = tin[(long long)gidx[i] * DH + f] * tanhf(vals[i]);
}

__global__ void k_att(const float* __restrict__ xn, const float* __restrict__ att,
                      float* __restrict__ a1, float* __restrict__ a2, int nrows) {
  int row = blockIdx.x * (blockDim.x >> 5) + (threadIdx.x >> 5);
  int lane = threadIdx.x & 31;
  if (row >= nrows) return;
  const float* x = xn + (long long)row * DH;
  float x0 = x[lane], x1 = x[lane + 32];
  float s1 = wave_sum(x0 * att[lane] + x1 * att[lane + 32]);
  float s2 = wave_sum(x0 * att[64 + lane] + x1 * att[96 + lane]);
  if (lane == 0) { a1[row] = s1; a2[row] = s2; }
}

__global__ void k_new_adj(const float* __restrict__ a1, const float* __restrict__ a2,
                          const float* __restrict__ adj_old, const int* __restrict__ gidx,
                          float* __restrict__ adj_new, int k, int npg_old, int total) {
  int idx = blockIdx.x * blockDim.x + threadIdx.x;
  if (idx >= total) return;
  int c = idx % k, r = (idx / k) % k, g = idx / (k * k);
  float e = a1[g * k + r] + a2[g * k + c];
  e = (e > 0.f) ? e : 0.2f * e;   // leaky_relu(., 0.2)
  e = fmaxf(e, 0.f);              // relu
  int lr = gidx[g * k + r] - g * npg_old;
  int lc = gidx[g * k + c] - g * npg_old;
  adj_new[((long long)g * k + r) * k + c] =
      e + adj_old[((long long)g * npg_old + lr) * npg_old + lc];   // LAMB = 1
}

// ------------------------------ readout / head -----------------------------
__global__ void k_readout(const float* __restrict__ t, float* __restrict__ out, int k) {
  int g = blockIdx.x, f = threadIdx.x;  // blockDim = 64
  const float* base = t + (long long)g * k * DH;
  float mx = -1e30f, sm = 0.f;
  for (int r = 0; r < k; ++r) {
    float v = base[(long long)r * DH + f];
    mx = fmaxf(mx, v); sm += v;
  }
  out[g * 128 + f] = mx;
  out[g * 128 + 64 + f] = sm / (float)k;
}

__global__ void k_head(const float* __restrict__ x1, const float* __restrict__ x2,
                       const float* __restrict__ x3,
                       const float* __restrict__ lw1, const float* __restrict__ lb1,
                       const float* __restrict__ lw2, const float* __restrict__ lb2,
                       const float* __restrict__ lw3, const float* __restrict__ lb3,
                       float* __restrict__ out) {
  __shared__ float r[128], h1[64], h2[32], o[8];
  int g = blockIdx.x, t = threadIdx.x;  // blockDim = 128
  r[t] = fmaxf(x1[g * 128 + t], 0.f) + fmaxf(x2[g * 128 + t], 0.f)
       + fmaxf(x3[g * 128 + t], 0.f);
  __syncthreads();
  if (t < 64) {
    float s = lb1[t];
    for (int j = 0; j < 128; ++j) s += lw1[t * 128 + j] * r[j];
    h1[t] = fmaxf(s, 0.f);
  }
  __syncthreads();
  if (t < 32) {
    float s = lb2[t];
    for (int j = 0; j < 64; ++j) s += lw2[t * 64 + j] * h1[j];
    h2[t] = fmaxf(s, 0.f);
  }
  __syncthreads();
  if (t < 6) {
    float s = lb3[t];
    for (int j = 0; j < 32; ++j) s += lw3[t * 32 + j] * h2[j];
    o[t] = s;
  }
  __syncthreads();
  if (t == 0) {
    float mx = -1e30f;
    for (int j = 0; j < 6; ++j) mx = fmaxf(mx, o[j]);
    float se = 0.f;
    for (int j = 0; j < 6; ++j) se += expf(o[j] - mx);
    float lse = mx + logf(se);
    for (int j = 0; j < 6; ++j) out[g * 6 + j] = o[j] - lse;
  }
}

// ---------------------------------------------------------------------------
extern "C" void kernel_launch(void* const* d_in, const int* in_sizes, int n_in,
                              void* d_out, int out_size, void* d_ws, size_t ws_size,
                              hipStream_t stream) {
  const float* x    = (const float*)d_in[0];
  const int*   ei   = (const int*)d_in[1];
  const float* W1   = (const float*)d_in[2];
  const float* b1   = (const float*)d_in[3];
  const float* W2   = (const float*)d_in[4];
  const float* b2   = (const float*)d_in[5];
  const float* W3   = (const float*)d_in[6];
  const float* b3   = (const float*)d_in[7];
  const float* att1 = (const float*)d_in[8];
  const float* att2 = (const float*)d_in[9];
  const float* lw1  = (const float*)d_in[10];
  const float* lb1  = (const float*)d_in[11];
  const float* lw2  = (const float*)d_in[12];
  const float* lb2  = (const float*)d_in[13];
  const float* lw3  = (const float*)d_in[14];
  const float* lb3  = (const float*)d_in[15];
  float* out = (float*)d_out;
  const int E = in_sizes[1] / 2;  // 131072 edges

  // ---- workspace carve (~30 MB; all buffers fully written before read) ----
  char* w = (char*)d_ws;
  auto carve = [&](size_t bytes) {
    void* p = (void*)w;
    w += (bytes + 255) & ~(size_t)255;
    return p;
  };
  float* adj1 = (float*)carve(sizeof(float) * NB * NPG0 * NPG0);  // 8 MB
  float* adj2 = (float*)carve(sizeof(float) * NB * 128 * 128);
  float* adj3 = (float*)carve(sizeof(float) * NB * 64 * 64);
  float* xh   = (float*)carve(sizeof(float) * NTOT * DIN);
  float* mx   = (float*)carve(sizeof(float) * NTOT * DH);
  float* hbuf = (float*)carve(sizeof(float) * NTOT * DH);
  float* tl   = (float*)carve(sizeof(float) * NTOT * DH);
  float* agg  = (float*)carve(sizeof(float) * NTOT * DH);
  float* t1   = (float*)carve(sizeof(float) * NTOT * DH);
  float* xs   = (float*)carve(sizeof(float) * NTOT * DH);
  float* prop = (float*)carve(sizeof(float) * NTOT * DH);
  float* tp1  = (float*)carve(sizeof(float) * NB * 128 * DH);
  float* tp2  = (float*)carve(sizeof(float) * NB * 64 * DH);
  float* dis  = (float*)carve(sizeof(float) * NTOT);
  float* scr  = (float*)carve(sizeof(float) * NTOT);
  int*   gidx = (int*)carve(sizeof(int) * NB * 128);
  float* vals = (float*)carve(sizeof(float) * NB * 128);
  float* a1   = (float*)carve(sizeof(float) * NB * 128);
  float* a2   = (float*)carve(sizeof(float) * NB * 128);
  float* hb   = (float*)carve(sizeof(float) * DH);
  float* x1r  = (float*)carve(sizeof(float) * NB * 128);
  float* x2r  = (float*)carve(sizeof(float) * NB * 128);
  float* x3r  = (float*)carve(sizeof(float) * NB * 128);
  (void)ws_size; (void)n_in; (void)out_size;

  // ---- adjacency 1 (block-diagonal scatter) ----
  hipMemsetAsync(adj1, 0, sizeof(float) * NB * NPG0 * NPG0, stream);
  k_build_adj<<<(E + 255) / 256, 256, 0, stream>>>(x, ei, adj1, E, NPG0, DIN);

  // ================= layer 1 (nr=8192, Din=128, npg=256) =================
  k_hyp_bias<<<1, 32, 0, stream>>>(b1, hb);
  k_expmap0_proj<<<NTOT / 8, 256, 0, stream>>>(x, xh, NTOT, DIN);
  wmma_gemm<true><<<dim3(NTOT / 64, DH / 16, 1), 128, 0, stream>>>(
      xh, W1, mx, NTOT, DH, DIN, 0, 0, 0);
  k_mobius<<<NTOT / 8, 256, 0, stream>>>(xh, mx, hb, hbuf, NTOT, DIN);
  k_logmap0<<<NTOT / 8, 256, 0, stream>>>(hbuf, tl, NTOT);
  wmma_gemm<false><<<dim3(NPG0 / 64, DH / 16, NB), 128, 0, stream>>>(
      adj1, tl, agg, NPG0, DH, NPG0,
      (long long)NPG0 * NPG0, (long long)NPG0 * DH, (long long)NPG0 * DH);
  k_expmap0_proj<<<NTOT / 8, 256, 0, stream>>>(agg, hbuf, NTOT, DH);
  k_logrelu<<<NTOT / 8, 256, 0, stream>>>(hbuf, t1, NTOT);

  // ---- pool 1: npg=256 -> k=128 ----
  k_degree<<<NTOT / 8, 256, 0, stream>>>(adj1, dis, NTOT, NPG0);
  k_scale_rows<<<NTOT * DH / 256, 256, 0, stream>>>(t1, dis, xs, NTOT * DH);
  wmma_gemm<false><<<dim3(NPG0 / 64, DH / 16, NB), 128, 0, stream>>>(
      adj1, xs, prop, NPG0, DH, NPG0,
      (long long)NPG0 * NPG0, (long long)NPG0 * DH, (long long)NPG0 * DH);
  k_score<<<NTOT / 8, 256, 0, stream>>>(t1, prop, dis, scr, NTOT);
  k_topk<<<NB, NPG0, 0, stream>>>(scr, gidx, vals, NPG0, 128);
  k_gather_scale<<<NB * 128 * DH / 256, 256, 0, stream>>>(t1, gidx, vals, tp1,
                                                          NB * 128 * DH);
  k_att<<<NB * 128 / 8, 256, 0, stream>>>(tp1, att1, a1, a2, NB * 128);
  k_new_adj<<<NB * 128 * 128 / 256, 256, 0, stream>>>(a1, a2, adj1, gidx, adj2,
                                                      128, NPG0, NB * 128 * 128);
  k_readout<<<NB, 64, 0, stream>>>(tp1, x1r, 128);

  // ================= layer 2 (nr=4096, Din=64, npg=128) =================
  const int N2 = NB * 128;
  k_hyp_bias<<<1, 32, 0, stream>>>(b2, hb);
  k_expmap0_proj<<<N2 / 8, 256, 0, stream>>>(tp1, xh, N2, DH);
  wmma_gemm<true><<<dim3(N2 / 64, DH / 16, 1), 128, 0, stream>>>(
      xh, W2, mx, N2, DH, DH, 0, 0, 0);
  k_mobius<<<N2 / 8, 256, 0, stream>>>(xh, mx, hb, hbuf, N2, DH);
  k_logmap0<<<N2 / 8, 256, 0, stream>>>(hbuf, tl, N2);
  wmma_gemm<false><<<dim3(128 / 64, DH / 16, NB), 128, 0, stream>>>(
      adj2, tl, agg, 128, DH, 128,
      (long long)128 * 128, (long long)128 * DH, (long long)128 * DH);
  k_expmap0_proj<<<N2 / 8, 256, 0, stream>>>(agg, hbuf, N2, DH);
  k_logrelu<<<N2 / 8, 256, 0, stream>>>(hbuf, t1, N2);

  // ---- pool 2: npg=128 -> k=64 ----
  k_degree<<<N2 / 8, 256, 0, stream>>>(adj2, dis, N2, 128);
  k_scale_rows<<<N2 * DH / 256, 256, 0, stream>>>(t1, dis, xs, N2 * DH);
  wmma_gemm<false><<<dim3(128 / 64, DH / 16, NB), 128, 0, stream>>>(
      adj2, xs, prop, 128, DH, 128,
      (long long)128 * 128, (long long)128 * DH, (long long)128 * DH);
  k_score<<<N2 / 8, 256, 0, stream>>>(t1, prop, dis, scr, N2);
  k_topk<<<NB, 128, 0, stream>>>(scr, gidx, vals, 128, 64);
  k_gather_scale<<<NB * 64 * DH / 256, 256, 0, stream>>>(t1, gidx, vals, tp2,
                                                         NB * 64 * DH);
  k_att<<<NB * 64 / 8, 256, 0, stream>>>(tp2, att2, a1, a2, NB * 64);
  k_new_adj<<<NB * 64 * 64 / 256, 256, 0, stream>>>(a1, a2, adj2, gidx, adj3,
                                                    64, 128, NB * 64 * 64);
  k_readout<<<NB, 64, 0, stream>>>(tp2, x2r, 64);

  // ================= layer 3 (nr=2048, Din=64, npg=64) =================
  const int N3 = NB * 64;
  k_hyp_bias<<<1, 32, 0, stream>>>(b3, hb);
  k_expmap0_proj<<<N3 / 8, 256, 0, stream>>>(tp2, xh, N3, DH);
  wmma_gemm<true><<<dim3(N3 / 64, DH / 16, 1), 128, 0, stream>>>(
      xh, W3, mx, N3, DH, DH, 0, 0, 0);
  k_mobius<<<N3 / 8, 256, 0, stream>>>(xh, mx, hb, hbuf, N3, DH);
  k_logmap0<<<N3 / 8, 256, 0, stream>>>(hbuf, tl, N3);
  wmma_gemm<false><<<dim3(64 / 64, DH / 16, NB), 128, 0, stream>>>(
      adj3, tl, agg, 64, DH, 64,
      (long long)64 * 64, (long long)64 * DH, (long long)64 * DH);
  k_expmap0_proj<<<N3 / 8, 256, 0, stream>>>(agg, hbuf, N3, DH);
  k_logrelu<<<N3 / 8, 256, 0, stream>>>(hbuf, t1, N3);
  k_readout<<<NB, 64, 0, stream>>>(t1, x3r, 64);

  // ================= head =================
  k_head<<<NB, 128, 0, stream>>>(x1r, x2r, x3r, lw1, lb1, lw2, lb2, lw3, lb3, out);
}
